// RMSLinearAttention_17222818857193
// MI455X (gfx1250) — matrix operand order repaired and checked by reading, not compiled, not run on hardware
//
#include <hip/hip_runtime.h>
#include <hip/hip_bf16.h>

typedef __attribute__((ext_vector_type(16))) _Float16 v16h;
typedef __attribute__((ext_vector_type(8)))  float    v8f;
typedef __attribute__((ext_vector_type(4)))  float    v4f;
typedef __attribute__((ext_vector_type(4)))  _Float16 v4h;
typedef __attribute__((ext_vector_type(2)))  _Float16 v2h;

union FragH { v16h h; uint4 u[2]; };

// ---------------------------------------------------------------------------
// CDNA5 async copy global->LDS (ASYNCcnt path) + wait
// ---------------------------------------------------------------------------
static __device__ __forceinline__ void async_b128(uint32_t lds_off, uint32_t gbyte_off,
                                                  const void* gbase) {
    asm volatile("global_load_async_to_lds_b128 %0, %1, %2"
                 :: "v"(lds_off), "v"(gbyte_off), "s"((uint64_t)(uintptr_t)gbase)
                 : "memory");
}
static __device__ __forceinline__ void wait_async0() {
    asm volatile("s_wait_asynccnt 0x0" ::: "memory");
}

// ---------------------------------------------------------------------------
// WMMA helper: D = A(16x32 f16) * B(32x16 f16) + C(16x16 f32)
// ---------------------------------------------------------------------------
static __device__ __forceinline__ v8f wmma_f16(v16h a, v16h b, v8f c) {
    return __builtin_amdgcn_wmma_f32_16x16x32_f16(
        /*neg_a=*/false, a, /*neg_b=*/false, b,
        /*c_mod=*/(short)0, c, /*reuse_a=*/false, /*reuse_b=*/false);
}

// A fragment (16x32, row-major, K contiguous). Per ISA 16-bit A layout:
// lanes 0-15 hold row m=lane, K={kb+0..7, kb+16..23}; lanes 16-31 hold
// K={kb+8..15, kb+24..31}. Two 16B loads per lane.
static __device__ __forceinline__ v16h load_fragA(const _Float16* __restrict__ A,
                                                  int lda, int row0, int kb, int lane) {
    int r  = row0 + (lane & 15);
    int kh = (lane >> 4) << 3;           // 0 or 8
    const _Float16* p = A + (size_t)r * lda + kb + kh;
    FragH f;
    f.u[0] = *reinterpret_cast<const uint4*>(p);
    f.u[1] = *reinterpret_cast<const uint4*>(p + 16);
    return f.h;
}

// B fragment from an LDS-staged weight tile. Tile layout: [col][COL_STRIDE f16],
// col c holds K=kb+0..31 contiguous. Lanes 0-15 (col n) read K {0..7, 16..23},
// lanes 16-31 read K {8..15, 24..31} -> two ds_load_b128.
#define COL_STRIDE 40              // 32 f16 payload + 8 f16 pad (80B: conflict-free)
static __device__ __forceinline__ v16h lds_fragB(const _Float16* tile, int col0, int lane) {
    int c  = col0 + (lane & 15);
    int ko = (lane >> 4) << 3;           // 0 or 8
    const _Float16* p = tile + c * COL_STRIDE + ko;
    FragH f;
    f.u[0] = *reinterpret_cast<const uint4*>(p);
    f.u[1] = *reinterpret_cast<const uint4*>(p + 16);
    return f.h;
}

// Issue async copy of one weight tile (NCOLS x 32 K, f16) into LDS.
// 16B chunk per lane: col = idx>>2, chunk = idx&3. 256 threads cooperate;
// NCOLS is a compile-time constant so this is fully unrolled and branch-free.
template <int NCOLS>
static __device__ __forceinline__ void issue_tile(const _Float16* W, int K, int nblk, int kb,
                                                  _Float16* ltile, int tid) {
#pragma unroll
    for (int base = 0; base < NCOLS * 4; base += 256) {
        const int idx = base + tid;
        const int col = idx >> 2, chunk = idx & 3;
        uint32_t go  = (uint32_t)(((nblk + col) * K + kb) * 2) + chunk * 16;
        uint32_t lds = (uint32_t)(uintptr_t)(ltile + col * COL_STRIDE) + chunk * 16;
        async_b128(lds, go, W);
    }
}

// ---------------------------------------------------------------------------
// f32 -> f16 conversion (vectorized x4, grid-stride)
// ---------------------------------------------------------------------------
__global__ void rla_cvt_f16(const float* __restrict__ in, _Float16* __restrict__ out, int n4) {
    int i      = blockIdx.x * blockDim.x + threadIdx.x;
    int stride = gridDim.x * blockDim.x;
    for (; i < n4; i += stride) {
        v4f x = reinterpret_cast<const v4f*>(in)[i];
        reinterpret_cast<v4h*>(out)[i] = __builtin_convertvector(x, v4h);
    }
}

// ---------------------------------------------------------------------------
// Fused GEMM: K = X @ Wk^T ; G = sigmoid(X @ Wkg^T + bkg)
// Block 256 thr = 8 waves (4 along M, 2 along N); block tile 128x64.
// Wave tile 32x32 = 2x2 WMMA tiles, dual accumulators (k and g).
// Weight tiles double-buffered in LDS via async copies; each stage holds a
// 64-deep K slice (2 x 32-K sub-tiles) -> 16 WMMAs per wave per barrier.
// ---------------------------------------------------------------------------
#define KG_NCOLS 64
#define KG_SUB   (KG_NCOLS * COL_STRIDE)      // f16 per 32-K sub-tile
#define KG_STAGE (4 * KG_SUB)                 // [mat][ksub] per buffer
__global__ __launch_bounds__(256)
void rla_gemm_kg(const _Float16* __restrict__ Xh,
                 const _Float16* __restrict__ Wkh,
                 const _Float16* __restrict__ Wkgh,
                 const float* __restrict__ bkg,
                 float* __restrict__ Kf, float* __restrict__ Gf,
                 int M, int N, int K) {
    __shared__ _Float16 smem[2 * KG_STAGE];   // [buf][mat][ksub][tile]
    const int tid  = threadIdx.x;
    const int lane = tid & 31;
    const int wave = tid >> 5;
    const int m0   = blockIdx.x * 128 + (wave >> 1) * 32;
    const int nblk = blockIdx.y * 64;
    const int n0l  = (wave & 1) * 32;         // wave's col base within block tile

    v8f ak[2][2], ag[2][2];
    const v8f vzero = {};
#pragma unroll
    for (int i = 0; i < 2; ++i)
#pragma unroll
        for (int j = 0; j < 2; ++j) { ak[i][j] = vzero; ag[i][j] = vzero; }

    // prologue: stage first 64-K slice of both weight matrices
    issue_tile<KG_NCOLS>(Wkh,  K, nblk, 0,  &smem[0 * KG_SUB], tid);
    issue_tile<KG_NCOLS>(Wkh,  K, nblk, 32, &smem[1 * KG_SUB], tid);
    issue_tile<KG_NCOLS>(Wkgh, K, nblk, 0,  &smem[2 * KG_SUB], tid);
    issue_tile<KG_NCOLS>(Wkgh, K, nblk, 32, &smem[3 * KG_SUB], tid);
    wait_async0();
    __syncthreads();

    for (int ks = 0; ks < K; ks += 64) {
        const int cur = (ks >> 6) & 1;
        _Float16* bufc = &smem[cur * KG_STAGE];
        if (ks + 64 < K) {
            _Float16* bufn = &smem[(1 - cur) * KG_STAGE];
            issue_tile<KG_NCOLS>(Wkh,  K, nblk, ks + 64, bufn + 0 * KG_SUB, tid);
            issue_tile<KG_NCOLS>(Wkh,  K, nblk, ks + 96, bufn + 1 * KG_SUB, tid);
            issue_tile<KG_NCOLS>(Wkgh, K, nblk, ks + 64, bufn + 2 * KG_SUB, tid);
            issue_tile<KG_NCOLS>(Wkgh, K, nblk, ks + 96, bufn + 3 * KG_SUB, tid);
        }
#pragma unroll
        for (int s = 0; s < 2; ++s) {
            const int kb = ks + s * 32;
            v16h a0  = load_fragA(Xh, K, m0,      kb, lane);
            v16h a1  = load_fragA(Xh, K, m0 + 16, kb, lane);
            v16h bk0 = lds_fragB(bufc + s * KG_SUB,            n0l,      lane);
            v16h bk1 = lds_fragB(bufc + s * KG_SUB,            n0l + 16, lane);
            v16h bg0 = lds_fragB(bufc + (2 + s) * KG_SUB,      n0l,      lane);
            v16h bg1 = lds_fragB(bufc + (2 + s) * KG_SUB,      n0l + 16, lane);

            ak[0][0] = wmma_f16(a0, bk0, ak[0][0]);
            ak[0][1] = wmma_f16(a0, bk1, ak[0][1]);
            ak[1][0] = wmma_f16(a1, bk0, ak[1][0]);
            ak[1][1] = wmma_f16(a1, bk1, ak[1][1]);
            ag[0][0] = wmma_f16(a0, bg0, ag[0][0]);
            ag[0][1] = wmma_f16(a0, bg1, ag[0][1]);
            ag[1][0] = wmma_f16(a1, bg0, ag[1][0]);
            ag[1][1] = wmma_f16(a1, bg1, ag[1][1]);
        }
        wait_async0();          // my async writes for next buffer landed
        __syncthreads();        // everyone done reading cur + all writes visible
    }

    // Epilogue. C layout: VGPR r, lane l -> row = r + (l<16?0:8), col = l&15.
    const int colb   = lane & 15;
    const int rowoff = (lane >> 4) << 3;
#pragma unroll
    for (int nt = 0; nt < 2; ++nt) {
        const int n    = nblk + n0l + nt * 16 + colb;
        const float bias = bkg[n];
#pragma unroll
        for (int mt = 0; mt < 2; ++mt) {
#pragma unroll
            for (int r = 0; r < 8; ++r) {
                const int m = m0 + mt * 16 + rowoff + r;
                const size_t idx = (size_t)m * N + n;
                Kf[idx] = ak[mt][nt][r];
                const float v = ag[mt][nt][r] + bias;
                Gf[idx] = 1.0f / (1.0f + __expf(-v));
            }
        }
    }
}

// ---------------------------------------------------------------------------
// GEMM: OUT = O @ Wout^T  (f16 inputs, f32 output)
// Block 256 thr = 8 waves (2 along M, 4 along N); block tile 128x128.
// Wave tile 64x32 = 4x2 WMMA tiles. Weight tiles double-buffered in LDS,
// 64-K slice per stage -> 16 WMMAs per wave per barrier.
// ---------------------------------------------------------------------------
#define OUT_NCOLS 128
#define OUT_SUB   (OUT_NCOLS * COL_STRIDE)
#define OUT_STAGE (2 * OUT_SUB)
__global__ __launch_bounds__(256)
void rla_gemm_out(const _Float16* __restrict__ Oh,
                  const _Float16* __restrict__ Wh,
                  float* __restrict__ Out,
                  int M, int N, int K) {
    __shared__ _Float16 smem[2 * OUT_STAGE];  // [buf][ksub][tile]
    const int tid  = threadIdx.x;
    const int lane = tid & 31;
    const int wave = tid >> 5;
    const int m0   = blockIdx.x * 128 + (wave >> 2) * 64;
    const int nblk = blockIdx.y * 128;
    const int n0l  = (wave & 3) * 32;

    v8f acc[4][2];
    const v8f vzero = {};
#pragma unroll
    for (int i = 0; i < 4; ++i)
#pragma unroll
        for (int j = 0; j < 2; ++j) acc[i][j] = vzero;

    issue_tile<OUT_NCOLS>(Wh, K, nblk, 0,  &smem[0],       tid);
    issue_tile<OUT_NCOLS>(Wh, K, nblk, 32, &smem[OUT_SUB], tid);
    wait_async0();
    __syncthreads();

    for (int ks = 0; ks < K; ks += 64) {
        const int cur = (ks >> 6) & 1;
        _Float16* bufc = &smem[cur * OUT_STAGE];
        if (ks + 64 < K) {
            _Float16* bufn = &smem[(1 - cur) * OUT_STAGE];
            issue_tile<OUT_NCOLS>(Wh, K, nblk, ks + 64, bufn,           tid);
            issue_tile<OUT_NCOLS>(Wh, K, nblk, ks + 96, bufn + OUT_SUB, tid);
        }
#pragma unroll
        for (int s = 0; s < 2; ++s) {
            const int kb = ks + s * 32;
            v16h a[4];
#pragma unroll
            for (int mt = 0; mt < 4; ++mt)
                a[mt] = load_fragA(Oh, K, m0 + mt * 16, kb, lane);
            v16h b0 = lds_fragB(bufc + s * OUT_SUB, n0l,      lane);
            v16h b1 = lds_fragB(bufc + s * OUT_SUB, n0l + 16, lane);
#pragma unroll
            for (int mt = 0; mt < 4; ++mt) {
                acc[mt][0] = wmma_f16(a[mt], b0, acc[mt][0]);
                acc[mt][1] = wmma_f16(a[mt], b1, acc[mt][1]);
            }
        }
        wait_async0();
        __syncthreads();
    }

    const int colb   = lane & 15;
    const int rowoff = (lane >> 4) << 3;
#pragma unroll
    for (int mt = 0; mt < 4; ++mt)
#pragma unroll
        for (int nt = 0; nt < 2; ++nt)
#pragma unroll
            for (int r = 0; r < 8; ++r) {
                const int m = m0 + mt * 16 + rowoff + r;
                const int n = nblk + n0l + nt * 16 + colb;
                Out[(size_t)m * N + n] = acc[mt][nt][r];
            }
}

// ---------------------------------------------------------------------------
// Sequential RMS-gated scan. One wave32 per (batch, head) sequence; 2 state
// elements per lane (d=64). In-wave butterfly reduction for sum(h^2).
// 16-deep register prefetch pipeline to hide L2 latency along the chain.
// ---------------------------------------------------------------------------
#define SCAN_PF 16
__global__ __launch_bounds__(32)
void rla_scan(const float* __restrict__ Kf, const float* __restrict__ Gf,
              _Float16* __restrict__ Oh, int B, int L, int D) {
    const int seq  = blockIdx.x;        // 0 .. B*16-1
    const int b    = seq >> 4;
    const int head = seq & 15;
    const int lane = threadIdx.x;

    const size_t base = ((size_t)b * L) * D + head * 64 + lane * 2;
    const float inv_sqrt_d = 0.125f;    // 1/sqrt(64)
    const float EPS = 1e-5f;

    float2 hx = make_float2(0.0f, 0.0f);
    float2 pk[SCAN_PF], pg[SCAN_PF];
#pragma unroll
    for (int i = 0; i < SCAN_PF; ++i) {
        pk[i] = *reinterpret_cast<const float2*>(Kf + base + (size_t)i * D);
        pg[i] = *reinterpret_cast<const float2*>(Gf + base + (size_t)i * D);
    }

    for (int t8 = 0; t8 < L; t8 += SCAN_PF) {
#pragma unroll
        for (int i = 0; i < SCAN_PF; ++i) {
            const int t = t8 + i;
            const float2 kv = pk[i];
            const float2 gv = pg[i];
            const int tp = t + SCAN_PF;
            if (tp < L) {
                pk[i] = *reinterpret_cast<const float2*>(Kf + base + (size_t)tp * D);
                pg[i] = *reinterpret_cast<const float2*>(Gf + base + (size_t)tp * D);
            }
            float ss = hx.x * hx.x + hx.y * hx.y;
#pragma unroll
            for (int m = 16; m >= 1; m >>= 1)
                ss += __shfl_xor(ss, m, 32);
            const float rms = __builtin_sqrtf(ss) * inv_sqrt_d;
            const float inv = 1.0f / (rms + EPS);
            hx.x = hx.x * inv * gv.x + kv.x;
            hx.y = hx.y * inv * gv.y + kv.y;
            v2h ov = { (_Float16)hx.x, (_Float16)hx.y };
            *reinterpret_cast<v2h*>(Oh + base + (size_t)t * D) = ov;
        }
    }
}

// ---------------------------------------------------------------------------
// Launch
// ---------------------------------------------------------------------------
extern "C" void kernel_launch(void* const* d_in, const int* in_sizes, int n_in,
                              void* d_out, int out_size, void* d_ws, size_t ws_size,
                              hipStream_t stream) {
    const float* x    = (const float*)d_in[0];
    const float* Wk   = (const float*)d_in[1];
    const float* Wkg  = (const float*)d_in[2];
    const float* bkg  = (const float*)d_in[3];
    const float* Wout = (const float*)d_in[4];
    (void)in_sizes; (void)n_in; (void)out_size; (void)ws_size;

    const int B = 8, L = 2048, D = 1024;
    const int M = B * L;                 // 16384

    char* ws = (char*)d_ws;
    size_t off = 0;
    auto carve = [&](size_t bytes) -> char* {
        char* p = ws + off;
        off += (bytes + 255) & ~(size_t)255;
        return p;
    };
    _Float16* xh    = (_Float16*)carve((size_t)M * D * sizeof(_Float16));
    _Float16* wkh   = (_Float16*)carve((size_t)D * D * sizeof(_Float16));
    _Float16* wkgh  = (_Float16*)carve((size_t)D * D * sizeof(_Float16));
    _Float16* wouth = (_Float16*)carve((size_t)D * D * sizeof(_Float16));
    float*    kf    = (float*)   carve((size_t)M * D * sizeof(float));
    float*    gf    = (float*)   carve((size_t)M * D * sizeof(float));
    _Float16* oh    = (_Float16*)carve((size_t)M * D * sizeof(_Float16));

    // 1) precision conversion
    rla_cvt_f16<<<4096, 256, 0, stream>>>(x,    xh,    M * D / 4);
    rla_cvt_f16<<<1024, 256, 0, stream>>>(Wk,   wkh,   D * D / 4);
    rla_cvt_f16<<<1024, 256, 0, stream>>>(Wkg,  wkgh,  D * D / 4);
    rla_cvt_f16<<<1024, 256, 0, stream>>>(Wout, wouth, D * D / 4);

    // 2) fused k / gate GEMM (block tile 128x64, 64-K stages)
    rla_gemm_kg<<<dim3(M / 128, D / 64), 256, 0, stream>>>(
        xh, wkh, wkgh, bkg, kf, gf, M, D, D);

    // 3) sequential RMS-gated scan: one wave per (batch, head)
    rla_scan<<<B * 16, 32, 0, stream>>>(kf, gf, oh, B, L, D);

    // 4) output projection (block tile 128x128, 64-K stages)
    rla_gemm_out<<<dim3(M / 128, D / 128), 256, 0, stream>>>(
        oh, wouth, (float*)d_out, M, D, D);
}
